// GRUClassifier_87557203296977
// MI455X (gfx1250) — matrix operand order repaired and checked by reading, not compiled, not run on hardware
//
#include <hip/hip_runtime.h>
#include <hip/hip_bf16.h>

// GRU classifier: I=128, H=512, C=1000, B=256, S=512.
// 16 persistent workgroups (16 batch rows each, 1024 threads = 32 waves) run
// the whole recurrence with h resident in LDS (bf16). Wave w owns hidden
// column tile w (16 cols) -> gate tiles {w, 32+w, 64+w}; only 4 v8f
// accumulators live per wave (no spills). Weights pre-packed to bf16 WMMA
// B-fragment layout, streamed from L2 each step. x tile per step prefetched
// by the Tensor Data Mover (double buffered).

#define I_DIM 128
#define H_DIM 512
#define C_DIM 1000
#define B_DIM 256
#define S_DIM 512

typedef __attribute__((ext_vector_type(16))) __bf16 v16bf;
typedef __attribute__((ext_vector_type(8)))  float  v8f;
typedef __attribute__((ext_vector_type(4))) unsigned int u32x4;
typedef __attribute__((ext_vector_type(8))) int i32x8;
typedef __attribute__((ext_vector_type(4))) int i32x4;

#if defined(__has_builtin)
#if __has_builtin(__builtin_amdgcn_tensor_load_to_lds) && __has_builtin(__builtin_amdgcn_s_wait_tensorcnt)
#define HAVE_TDM 1
#endif
#endif

__device__ __forceinline__ unsigned short f2bf(float f) {
  __bf16 b = (__bf16)f;
  return __builtin_bit_cast(unsigned short, b);
}
__device__ __forceinline__ float bf2f(unsigned short u) {
  __bf16 b = __builtin_bit_cast(__bf16, u);
  return (float)b;
}

union FragU { uint4 q[2]; v16bf v; };

// B fragment (K=32 x N=16 bf16): packed so each lane reads 32 contiguous bytes.
__device__ __forceinline__ v16bf load_frag_b(const unsigned short* p) {
  FragU f;
  f.q[0] = *(const uint4*)(p);
  f.q[1] = *(const uint4*)(p + 8);
  return f.v;
}
// A fragment (M=16 x K=32 bf16) from a row-major bf16 buffer.
// Lane L: M = L%16; elems 0..7 = K k0..k0+7, elems 8..15 = K k0+16..k0+23,
// where k0 = kt*32 + (L/16)*8  (matches ISA 16-bit A layout).
__device__ __forceinline__ v16bf load_frag_a(const unsigned short* row, int k0) {
  FragU f;
  f.q[0] = *(const uint4*)(row + k0);
  f.q[1] = *(const uint4*)(row + k0 + 16);
  return f.v;
}
__device__ __forceinline__ v8f wmma_bf16(v16bf a, v16bf b, v8f c) {
  return __builtin_amdgcn_wmma_f32_16x16x32_bf16(false, a, false, b, (short)0, c,
                                                 false, false);
}

// ---------------------------------------------------------------------------
// Pack f32 row-major W[G][K] into bf16 B-fragment blob [nt][kt][lane][16].
// Fragment element (lane, e): N = nt*16 + lane%16, K = kt*32 + (lane/16)*8 + e (e<8), kt*32 + 16 + (lane/16)*8 + e-8 (e>=8).
// (B[k][n] = W[n][k] since gemms are h @ W^T.)
// ---------------------------------------------------------------------------
__global__ void pack_kernel(const float* __restrict__ W, int Grows, int Kdim,
                            int KT, int totalElems, unsigned short* __restrict__ dst) {
  for (int idx = blockIdx.x * blockDim.x + threadIdx.x; idx < totalElems;
       idx += gridDim.x * blockDim.x) {
    int e    = idx & 15;
    int lane = (idx >> 4) & 31;
    int frag = idx >> 9;
    int kt   = frag % KT;
    int nt   = frag / KT;
    int g    = nt * 16 + (lane & 15);
    int k    = kt * 32 + (lane >> 4) * 8 + (e < 8 ? e : 8 + e);
    float v  = (g < Grows) ? W[(unsigned)g * Kdim + k] : 0.f;
    dst[idx] = f2bf(v);
  }
}

// ---------------------------------------------------------------------------
// Persistent GRU recurrence. grid = 16 blocks x 1024 threads (32 waves).
// Block owns batch rows [b0, b0+16). Wave w owns hidden-column tile w.
// ---------------------------------------------------------------------------
__global__ __launch_bounds__(1024) void gru_kernel(
    const float* __restrict__ x,
    const unsigned short* __restrict__ Wih_p,   // [96 nt][4 kt][32][16]
    const unsigned short* __restrict__ Whh_p,   // [96 nt][16 kt][32][16]
    const float* __restrict__ b_ih,
    const float* __restrict__ b_hh,
    unsigned short* __restrict__ hT)            // [256][512] bf16 out
{
  __shared__ unsigned short h_lds[16][520];     // padded rows, 16B-aligned
  __shared__ unsigned short x_bf[16][136];
#ifdef HAVE_TDM
  __shared__ float xstage[2][16 * 128];
#endif
  const int tid  = threadIdx.x;
  const int lane = tid & 31;
  const int wid  = tid >> 5;                    // 0..31 = hidden-col tile
  const int b0   = blockIdx.x * 16;
  const int mrow = lane & 15;                   // A-row index / D-col index
  const int hi   = lane >> 4;

  for (int i = tid; i < 16 * 520; i += 1024) ((unsigned short*)h_lds)[i] = 0;

  // Per-lane column biases (added in the gate epilogue, not to C).
  const int jcol = wid * 16 + mrow;             // column within H
  const float bias_r  = b_ih[jcol]             + b_hh[jcol];
  const float bias_z  = b_ih[H_DIM + jcol]     + b_hh[H_DIM + jcol];
  const float bias_xn = b_ih[2 * H_DIM + jcol];
  const float bias_hn = b_hh[2 * H_DIM + jcol];

  // Base pointers for this wave's three gate tiles (r: nt=wid, z: 32+wid,
  // n: 64+wid) in the packed fragment blobs.
  const unsigned laneOff = (unsigned)lane * 16u;
  const unsigned short* ih_r = Wih_p + ((unsigned)(wid)      * 4u) * 512u + laneOff;
  const unsigned short* ih_z = Wih_p + ((unsigned)(32 + wid) * 4u) * 512u + laneOff;
  const unsigned short* ih_n = Wih_p + ((unsigned)(64 + wid) * 4u) * 512u + laneOff;
  const unsigned short* hh_r = Whh_p + ((unsigned)(wid)      * 16u) * 512u + laneOff;
  const unsigned short* hh_z = Whh_p + ((unsigned)(32 + wid) * 16u) * 512u + laneOff;
  const unsigned short* hh_n = Whh_p + ((unsigned)(64 + wid) * 16u) * 512u + laneOff;

#ifdef HAVE_TDM
  // TDM descriptor: 2D tile [16 rows][128 f32], row stride S*I elements.
  auto tdm_issue = [&](int t, int buf) {
    unsigned long long gaddr =
        (unsigned long long)(const void*)(x + ((unsigned long long)b0 * S_DIM + t) * I_DIM);
    unsigned ldsOff = (unsigned)(unsigned long long)(&xstage[buf][0]);
    u32x4 g0;
    g0.x = 1u;                                        // count=1, no gather
    g0.y = ldsOff;                                    // lds_addr (bytes)
    g0.z = (unsigned)(gaddr & 0xFFFFFFFFull);         // global_addr[31:0]
    g0.w = (unsigned)((gaddr >> 32) & 0x1FFFFFFull) | (2u << 30);  // addr hi + type=2
    i32x8 g1;
    g1[0] = (int)(2u << 16);          // data_size = 4 bytes
    g1[1] = (int)(128u << 16);        // tensor_dim0 = 128 (bits 79:48)
    g1[2] = (int)(16u << 16);         // tensor_dim1 = 16  (bits 111:80)
    g1[3] = (int)(128u << 16);        // tile_dim0 = 128   (bits 127:112)
    g1[4] = 16;                       // tile_dim1 = 16    (bits 143:128)
    g1[5] = (int)(S_DIM * I_DIM);     // tensor_dim0_stride (bits 207:160)
    g1[6] = 0;
    g1[7] = 0;
    i32x4 gz4;
    gz4[0] = 0; gz4[1] = 0; gz4[2] = 0; gz4[3] = 0;
    i32x8 gz8;
    gz8[0] = 0; gz8[1] = 0; gz8[2] = 0; gz8[3] = 0;
    gz8[4] = 0; gz8[5] = 0; gz8[6] = 0; gz8[7] = 0;
    __builtin_amdgcn_tensor_load_to_lds(g0, g1, gz4, gz4, gz8, 0);
  };
  if (wid == 0) tdm_issue(0, 0);
#endif
  __syncthreads();

  for (int t = 0; t < S_DIM; ++t) {
#ifdef HAVE_TDM
    if (wid == 0) {
      if (t + 1 < S_DIM) {
        tdm_issue(t + 1, (t + 1) & 1);                 // prefetch next step
        __builtin_amdgcn_s_wait_tensorcnt(1);          // wait for step t's tile
      } else {
        __builtin_amdgcn_s_wait_tensorcnt(0);
      }
    }
    __syncthreads();
    {
      const float* st = &xstage[t & 1][0];
      for (int i = tid; i < 16 * 128; i += 1024)
        x_bf[i >> 7][i & 127] = f2bf(st[i]);
    }
#else
    for (int i = tid; i < 16 * 128; i += 1024) {
      int m = i >> 7, c = i & 127;
      x_bf[m][c] = f2bf(x[(((unsigned long long)(b0 + m)) * S_DIM + t) * I_DIM + c]);
    }
#endif
    __syncthreads();

    v8f acc_r  = {};
    v8f acc_z  = {};
    v8f acc_xn = {};
    v8f acc_hn = {};

    // x-part: K = 128 -> 4 k-tiles. A fragment shared by all 3 gate tiles.
#pragma unroll 1
    for (int kt = 0; kt < 4; ++kt) {
      v16bf a = load_frag_a(&x_bf[mrow][0], kt * 32 + hi * 8);
      unsigned o = (unsigned)kt * 512u;
      acc_r  = wmma_bf16(a, load_frag_b(ih_r + o), acc_r);
      acc_z  = wmma_bf16(a, load_frag_b(ih_z + o), acc_z);
      acc_xn = wmma_bf16(a, load_frag_b(ih_n + o), acc_xn);
    }
    // h-part: K = 512 -> 16 k-tiles, streamed from L2-resident packed W_hh.
#pragma unroll 1
    for (int kt = 0; kt < 16; ++kt) {
      v16bf a = load_frag_a(&h_lds[mrow][0], kt * 32 + hi * 8);
      unsigned o = (unsigned)kt * 512u;
      acc_r  = wmma_bf16(a, load_frag_b(hh_r + o), acc_r);
      acc_z  = wmma_bf16(a, load_frag_b(hh_z + o), acc_z);
      acc_hn = wmma_bf16(a, load_frag_b(hh_n + o), acc_hn);
    }

    __syncthreads();  // all waves done reading h as A fragments

    // Gate math + h update. D element v of lane: (M = hi*8+v, N = mrow).
    // Each wave reads/writes only its own 16-column tile -> no cross-wave
    // hazard inside this phase.
#pragma unroll
    for (int v = 0; v < 8; ++v) {
      int m = hi * 8 + v;
      float r   = 1.f / (1.f + __expf(-(acc_r[v] + bias_r)));
      float z   = 1.f / (1.f + __expf(-(acc_z[v] + bias_z)));
      float pre = (acc_xn[v] + bias_xn) + r * (acc_hn[v] + bias_hn);
      float n   = 2.f / (1.f + __expf(-2.f * pre)) - 1.f;   // tanh
      float hold = bf2f(h_lds[m][jcol]);
      h_lds[m][jcol] = f2bf((1.f - z) * n + z * hold);
    }
    __syncthreads();
  }

  // Final hidden state to workspace (bf16 row-major [B][H]).
  for (int i = tid; i < 16 * 512; i += 1024) {
    int m = i >> 9, k = i & 511;
    hT[(unsigned)(b0 + m) * 512 + k] = h_lds[m][k];
  }
}

// ---------------------------------------------------------------------------
// Head GEMM: out[256][1000] = hT @ W_head^T + b_head. One wave per 16x16 tile.
// ---------------------------------------------------------------------------
__global__ __launch_bounds__(256) void head_kernel(
    const unsigned short* __restrict__ hT,
    const unsigned short* __restrict__ Whead_p,   // [63 nt][16 kt][32][16]
    const float* __restrict__ b_head,
    float* __restrict__ out)
{
  const int tid = threadIdx.x, lane = tid & 31, wid = tid >> 5;
  const int NTILES = 63;                           // 1008 cols, guard to 1000
  const int tile = blockIdx.x * 8 + wid;
  if (tile >= 16 * NTILES) return;
  const int mt = tile / NTILES, nt = tile % NTILES;
  const int mrow = lane & 15, hi = lane >> 4;
  const int col = nt * 16 + mrow;

  v8f acc = {};

  const unsigned short* arow = hT + (unsigned)(mt * 16 + mrow) * 512;
  const unsigned short* bbase = Whead_p + (unsigned)(nt * 16) * 512u + (unsigned)lane * 16u;
#pragma unroll 1
  for (int kt = 0; kt < 16; ++kt) {
    v16bf a = load_frag_a(arow, kt * 32 + hi * 8);
    v16bf b = load_frag_b(bbase + (unsigned)kt * 512u);
    acc = wmma_bf16(a, b, acc);
  }
  if (col < C_DIM) {
    float bv = b_head[col];
#pragma unroll
    for (int v = 0; v < 8; ++v) {
      int m = hi * 8 + v;
      out[(unsigned)(mt * 16 + m) * C_DIM + col] = acc[v] + bv;
    }
  }
}

// ---------------------------------------------------------------------------
extern "C" void kernel_launch(void* const* d_in, const int* in_sizes, int n_in,
                              void* d_out, int out_size, void* d_ws, size_t ws_size,
                              hipStream_t stream) {
  const float* x      = (const float*)d_in[0];
  const float* W_ih   = (const float*)d_in[1];
  const float* W_hh   = (const float*)d_in[2];
  const float* b_ih   = (const float*)d_in[3];
  const float* b_hh   = (const float*)d_in[4];
  const float* W_head = (const float*)d_in[5];
  const float* b_head = (const float*)d_in[6];
  float* out = (float*)d_out;

  // Workspace layout (bf16 elements): ~3.1 MB total.
  unsigned short* ws      = (unsigned short*)d_ws;
  unsigned short* Wih_p   = ws;                       // 96*4*512
  unsigned short* Whh_p   = Wih_p + 96 * 4 * 512;     // 96*16*512
  unsigned short* Whead_p = Whh_p + 96 * 16 * 512;    // 63*16*512
  unsigned short* hT      = Whead_p + 63 * 16 * 512;  // 256*512

  pack_kernel<<<256, 256, 0, stream>>>(W_ih,   1536, 128, 4,  96 * 4 * 512,  Wih_p);
  pack_kernel<<<256, 256, 0, stream>>>(W_hh,   1536, 512, 16, 96 * 16 * 512, Whh_p);
  pack_kernel<<<256, 256, 0, stream>>>(W_head, 1000, 512, 16, 63 * 16 * 512, Whead_p);

  gru_kernel<<<16, 1024, 0, stream>>>(x, Wih_p, Whh_p, b_ih, b_hh, hT);
  head_kernel<<<126, 256, 0, stream>>>(hT, Whead_p, b_head, out);
}
